// LCPNClassificationHead_13950053778192
// MI455X (gfx1250) — matrix operand order repaired and checked by reading, not compiled, not run on hardware
//
#include <hip/hip_runtime.h>

// ---------------------------------------------------------------------------
// CDNA5 / gfx1250 (MI455X, wave32).  bf16x3-split WMMA GEMM + fused LCPN head.
// ---------------------------------------------------------------------------

typedef __attribute__((ext_vector_type(16))) __bf16 v16bf;
typedef __attribute__((ext_vector_type(8)))  float  v8f;

#define B_ROWS 16384
#define H_DIM  1024
#define MBLK   64     // rows per workgroup
#define KC     128    // K-chunk staged in LDS
#define LDA    136    // LDS row stride (elements): 272B -> 4-bank skew, 16B aligned
#define NTILES 71     // 63 (lvl2, N=1000) + 7 (lvl1, N=100) + 1 (lvl0, N=10)
#define ROUNDS 9      // ceil(71 / 8 waves)
#define NEG_INF (-3.0e38f)

static __device__ __forceinline__ unsigned short f2bf(float f) {
    unsigned int u = __float_as_uint(f);
    u += 0x7FFFu + ((u >> 16) & 1u);       // round-to-nearest-even
    return (unsigned short)(u >> 16);
}
static __device__ __forceinline__ float bf2f(unsigned short h) {
    return __uint_as_float(((unsigned int)h) << 16);
}

union Frag { uint4 u[2]; v16bf v; };       // 32B = 8 VGPRs = one 16-bit WMMA operand

static __device__ __forceinline__ v8f wmma_bf16(v16bf a, v16bf b, v8f c) {
    return __builtin_amdgcn_wmma_f32_16x16x32_bf16(
        /*neg_a=*/false, a, /*neg_b=*/false, b,
        /*c_mod=*/(short)0, c, /*reuse_a=*/false, /*reuse_b=*/false);
}

// ---------------------------------------------------------------------------
// Kernel 1: convert W0/W1/W2 (fp32) -> bf16 hi/lo planes, padded to 16-row tiles
// padded rows: lvl2 1008, lvl1 112, lvl0 16  (pad rows = 0)
// ---------------------------------------------------------------------------
__global__ __launch_bounds__(256) void convw_kernel(
    const float* __restrict__ W0, const float* __restrict__ W1,
    const float* __restrict__ W2,
    unsigned short* __restrict__ Whi2, unsigned short* __restrict__ Wlo2,
    unsigned short* __restrict__ Whi1, unsigned short* __restrict__ Wlo1,
    unsigned short* __restrict__ Whi0, unsigned short* __restrict__ Wlo0)
{
    int idx = blockIdx.x * 256 + threadIdx.x;      // one thread per 8 elements
    if (idx >= 1136 * (H_DIM / 8)) return;
    int r = idx >> 7;                              // padded row
    int k = (idx & 127) << 3;                      // column base
    const float* src = nullptr;
    unsigned short *dh, *dl;
    if (r < 1008) {
        if (r < 1000) src = W2 + (size_t)r * H_DIM + k;
        dh = Whi2 + (size_t)r * H_DIM + k;  dl = Wlo2 + (size_t)r * H_DIM + k;
    } else if (r < 1120) {
        int lr = r - 1008;
        if (lr < 100) src = W1 + (size_t)lr * H_DIM + k;
        dh = Whi1 + (size_t)lr * H_DIM + k; dl = Wlo1 + (size_t)lr * H_DIM + k;
    } else {
        int lr = r - 1120;
        if (lr < 10) src = W0 + (size_t)lr * H_DIM + k;
        dh = Whi0 + (size_t)lr * H_DIM + k; dl = Wlo0 + (size_t)lr * H_DIM + k;
    }
#pragma unroll
    for (int j = 0; j < 8; ++j) {
        float f = src ? src[j] : 0.0f;
        unsigned short h = f2bf(f);
        dh[j] = h;
        dl[j] = f2bf(f - bf2f(h));
    }
}

// ---------------------------------------------------------------------------
// Kernel 2: fused 3-level GEMM  logits = x @ W^T + b  via bf16x3 WMMA
//   grid = 256 WGs (64-row blocks), 256 threads = 8 waves
//   per round: wave w takes virtual N-tile t = round*8 + w (16 cols)
//   LDS: x block staged per 128-K chunk as bf16 hi/lo (inline fp32->bf16 split)
// ---------------------------------------------------------------------------
__global__ __launch_bounds__(256) void gemm_kernel(
    const float* __restrict__ x,
    const unsigned short* __restrict__ Whi2, const unsigned short* __restrict__ Wlo2,
    const unsigned short* __restrict__ Whi1, const unsigned short* __restrict__ Wlo1,
    const unsigned short* __restrict__ Whi0, const unsigned short* __restrict__ Wlo0,
    const float* __restrict__ b0, const float* __restrict__ b1,
    const float* __restrict__ b2,
    float* __restrict__ dout,            // lvl2 logits -> probs region [B,1000]
    float* __restrict__ lg1,             // [B,100]
    float* __restrict__ lg0)             // [B,10]
{
    __shared__ unsigned short Ahi[MBLK * LDA];
    __shared__ unsigned short Alo[MBLK * LDA];

    const int tid  = threadIdx.x;
    const int wave = tid >> 5;
    const int lane = tid & 31;
    const int m0   = blockIdx.x * MBLK;
    const int halfsel = lane & 16;       // 0 for lanes 0-15, 16 for lanes 16-31

    for (int round = 0; round < ROUNDS; ++round) {
        const int t = round * 8 + wave;
        bool active = (t < NTILES);
        const unsigned short *Bh = nullptr, *Bl = nullptr;
        const float* bias = nullptr;
        float* out = nullptr;
        int n0 = 0, Nl = 0, ostride = 0;
        if (t < 63)      { Bh = Whi2; Bl = Wlo2; bias = b2; out = dout;
                           n0 = t * 16;        Nl = 1000; ostride = 1000; }
        else if (t < 70) { Bh = Whi1; Bl = Wlo1; bias = b1; out = lg1;
                           n0 = (t - 63) * 16; Nl = 100;  ostride = 100; }
        else if (t == 70){ Bh = Whi0; Bl = Wlo0; bias = b0; out = lg0;
                           n0 = 0;             Nl = 10;   ostride = 10; }

        v8f acc[4] = {v8f{}, v8f{}, v8f{}, v8f{}};

        for (int kc = 0; kc < H_DIM; kc += KC) {
            __syncthreads();                       // protect LDS from prior readers
            // ---- cooperative stage: x fp32 -> bf16 hi/lo in LDS (64 x 128) ----
#pragma unroll
            for (int it = 0; it < 8; ++it) {
                int o   = tid + it * 256;          // quad id 0..2047
                int row = o >> 5;
                int q   = o & 31;
                const float4 vv = *(const float4*)(x + (size_t)(m0 + row) * H_DIM
                                                     + kc + (q << 2));
                float f[4] = {vv.x, vv.y, vv.z, vv.w};
                union { unsigned short s[4]; uint2 u; } ph, pl;
#pragma unroll
                for (int j = 0; j < 4; ++j) {
                    unsigned short h = f2bf(f[j]);
                    ph.s[j] = h;
                    pl.s[j] = f2bf(f[j] - bf2f(h));
                }
                *(uint2*)&Ahi[row * LDA + (q << 2)] = ph.u;
                *(uint2*)&Alo[row * LDA + (q << 2)] = pl.u;
            }
            __syncthreads();

            if (active) {
#pragma unroll
                for (int ks = 0; ks < KC; ks += 32) {
                    // B fragment (16 consecutive K per lane, hi & lo planes)
                    const int kglob = kc + ks + (halfsel ? 16 : 0);
                    const size_t boff = (size_t)(n0 + (lane & 15)) * H_DIM + kglob;
                    Frag Bhf, Blf;
                    const uint4* bph = (const uint4*)(Bh + boff);
                    const uint4* bpl = (const uint4*)(Bl + boff);
                    Bhf.u[0] = bph[0]; Bhf.u[1] = bph[1];
                    Blf.u[0] = bpl[0]; Blf.u[1] = bpl[1];

                    const int a_off = ks + (halfsel ? 16 : 0);
#pragma unroll
                    for (int st = 0; st < 4; ++st) {
                        const int r = st * 16 + (lane & 15);
                        Frag Ah, Al;
                        Ah.u[0] = *(const uint4*)&Ahi[r * LDA + a_off];
                        Ah.u[1] = *(const uint4*)&Ahi[r * LDA + a_off + 8];
                        Al.u[0] = *(const uint4*)&Alo[r * LDA + a_off];
                        Al.u[1] = *(const uint4*)&Alo[r * LDA + a_off + 8];
                        acc[st] = wmma_bf16(Ah.v, Bhf.v, acc[st]);  // hi*hi
                        acc[st] = wmma_bf16(Ah.v, Blf.v, acc[st]);  // hi*lo
                        acc[st] = wmma_bf16(Al.v, Bhf.v, acc[st]);  // lo*hi
                    }
                }
            }
        }

        if (active) {
            const int col = n0 + (lane & 15);
            if (col < Nl) {
                const float bv = bias[col];
                const int rbase = m0 + (halfsel ? 8 : 0);
#pragma unroll
                for (int st = 0; st < 4; ++st)
#pragma unroll
                    for (int v = 0; v < 8; ++v) {
                        int row = rbase + st * 16 + v;
                        out[(size_t)row * ostride + col] = acc[st][v] + bv;
                    }
            }
        }
    }
}

// ---------------------------------------------------------------------------
// Kernel 3: fused epilogue, one wave32 per batch row
//   - CE terms (logsumexp + target gather) for all 3 levels
//   - probs = sig(lg0[path0]) * sig(lg1[path1]) * sig(lg2)  (in-place in d_out)
//   - CE over probs; per-row loss -> ws
// ---------------------------------------------------------------------------
static __device__ __forceinline__ float wmaxf(float v) {
#pragma unroll
    for (int o = 16; o; o >>= 1) v = fmaxf(v, __shfl_xor(v, o, 32));
    return v;
}
static __device__ __forceinline__ float wsumf(float v) {
#pragma unroll
    for (int o = 16; o; o >>= 1) v += __shfl_xor(v, o, 32);
    return v;
}

__global__ __launch_bounds__(256) void head_kernel(
    const int* __restrict__ labels, const int* __restrict__ paths,
    const float* __restrict__ lg0, const float* __restrict__ lg1,
    float* __restrict__ dout, float* __restrict__ rowloss)
{
    __shared__ float sig0s[8][16];
    __shared__ float sig1s[8][112];

    const int wave = threadIdx.x >> 5;
    const int lane = threadIdx.x & 31;
    const int row  = blockIdx.x * 8 + wave;

    const int lab  = labels[row];
    const int lab0 = paths[lab * 3 + 0];
    const int lab1 = paths[lab * 3 + 1];
    const int lab2 = paths[lab * 3 + 2];

    // ---- level 0 (N=10) ----
    float v0 = (lane < 10) ? lg0[(size_t)row * 10 + lane] : NEG_INF;
    float m0 = wmaxf(v0);
    float s0 = wsumf((lane < 10) ? __expf(v0 - m0) : 0.0f);
    float lse0 = m0 + __logf(s0);
    float t0 = wsumf((lane == lab0) ? v0 : 0.0f);
    if (lane < 10) sig0s[wave][lane] = 1.0f / (1.0f + __expf(-v0));

    // ---- level 1 (N=100) ----
    float v1[4];
#pragma unroll
    for (int i = 0; i < 4; ++i) {
        int l = lane + 32 * i;
        v1[i] = (l < 100) ? lg1[(size_t)row * 100 + l] : NEG_INF;
    }
    float m1l = fmaxf(fmaxf(v1[0], v1[1]), fmaxf(v1[2], v1[3]));
    float m1 = wmaxf(m1l);
    float s1l = 0.0f, t1l = 0.0f;
#pragma unroll
    for (int i = 0; i < 4; ++i) {
        int l = lane + 32 * i;
        if (l < 100) {
            s1l += __expf(v1[i] - m1);
            if (l == lab1) t1l = v1[i];
            sig1s[wave][l] = 1.0f / (1.0f + __expf(-v1[i]));
        }
    }
    float lse1 = m1 + __logf(wsumf(s1l));
    float t1 = wsumf(t1l);

    __syncthreads();   // sig LDS visible to all lanes of this wave (uniform across WG)

    // ---- level 2 (N=1000) + probs ----
    float v2[32];
    float m2l = NEG_INF;
#pragma unroll
    for (int i = 0; i < 32; ++i) {
        int l = 32 * i + lane;
        v2[i] = (l < 1000) ? dout[(size_t)row * 1000 + l] : NEG_INF;
        m2l = fmaxf(m2l, v2[i]);
    }
    float m2 = wmaxf(m2l);
    float s2l = 0.0f, t2l = 0.0f;
#pragma unroll
    for (int i = 0; i < 32; ++i) {
        int l = 32 * i + lane;
        if (l < 1000) {
            s2l += __expf(v2[i] - m2);
            if (l == lab2) t2l = v2[i];
        }
    }
    float lse2 = m2 + __logf(wsumf(s2l));
    float t2 = wsumf(t2l);

    float p[32];
    float pml = NEG_INF;
#pragma unroll
    for (int i = 0; i < 32; ++i) {
        int l = 32 * i + lane;
        if (l < 1000) {
            float sg2 = 1.0f / (1.0f + __expf(-v2[i]));
            float pp  = sg2 * sig1s[wave][l / 10] * sig0s[wave][l / 100];
            p[i] = pp;
            dout[(size_t)row * 1000 + l] = pp;     // overwrite logits with probs
            pml = fmaxf(pml, pp);
        } else p[i] = NEG_INF;
    }
    float pm = wmaxf(pml);
    float psl = 0.0f, tpl = 0.0f;
#pragma unroll
    for (int i = 0; i < 32; ++i) {
        int l = 32 * i + lane;
        if (l < 1000) {
            psl += __expf(p[i] - pm);
            if (l == lab2) tpl = p[i];
        }
    }
    float lsep = pm + __logf(wsumf(psl));
    float tp = wsumf(tpl);

    if (lane == 0)
        rowloss[row] = (lse0 - t0) + (lse1 - t1) + (lse2 - t2) + (lsep - tp);
}

// ---------------------------------------------------------------------------
// Kernel 4: deterministic reduction of per-row losses (fixed-order tree)
// ---------------------------------------------------------------------------
__global__ __launch_bounds__(256) void reduce_kernel(
    const float* __restrict__ rowloss, float* __restrict__ dout)
{
    __shared__ float red[256];
    int t = threadIdx.x;
    float s = 0.0f;
    for (int i = t; i < B_ROWS; i += 256) s += rowloss[i];
    red[t] = s;
    __syncthreads();
    for (int o = 128; o; o >>= 1) {
        if (t < o) red[t] += red[t + o];
        __syncthreads();
    }
    if (t == 0) dout[(size_t)B_ROWS * 1000] = red[0] * (1.0f / (float)B_ROWS);
}

// ---------------------------------------------------------------------------
extern "C" void kernel_launch(void* const* d_in, const int* in_sizes, int n_in,
                              void* d_out, int out_size, void* d_ws, size_t ws_size,
                              hipStream_t stream) {
    (void)in_sizes; (void)n_in; (void)out_size; (void)ws_size;
    const float* x      = (const float*)d_in[0];
    const int*   labels = (const int*)  d_in[1];
    const float* W0     = (const float*)d_in[2];
    const float* b0     = (const float*)d_in[3];
    const float* W1     = (const float*)d_in[4];
    const float* b1     = (const float*)d_in[5];
    const float* W2     = (const float*)d_in[6];
    const float* b2     = (const float*)d_in[7];
    const int*   paths  = (const int*)  d_in[8];
    float* dout = (float*)d_out;

    // workspace layout (bytes)
    char* w = (char*)d_ws;
    unsigned short* Whi2 = (unsigned short*)(w);                       // 1008*1024*2
    unsigned short* Wlo2 = (unsigned short*)(w + 2064384);
    unsigned short* Whi1 = (unsigned short*)(w + 4128768);             // 112*1024*2
    unsigned short* Wlo1 = (unsigned short*)(w + 4358144);
    unsigned short* Whi0 = (unsigned short*)(w + 4587520);             // 16*1024*2
    unsigned short* Wlo0 = (unsigned short*)(w + 4620288);
    float* lg1     = (float*)(w + 4653056);                            // 16384*100*4
    float* lg0     = (float*)(w + 11206656);                           // 16384*10*4
    float* rowloss = (float*)(w + 11862016);                           // 16384*4

    convw_kernel<<<568, 256, 0, stream>>>(W0, W1, W2,
                                          Whi2, Wlo2, Whi1, Wlo1, Whi0, Wlo0);
    gemm_kernel<<<B_ROWS / MBLK, 256, 0, stream>>>(x,
        Whi2, Wlo2, Whi1, Wlo1, Whi0, Wlo0, b0, b1, b2, dout, lg1, lg0);
    head_kernel<<<B_ROWS / 8, 256, 0, stream>>>(labels, paths, lg0, lg1,
                                                dout, rowloss);
    reduce_kernel<<<1, 256, 0, stream>>>(rowloss, dout);
}
